// Decoder_27882927686155
// MI455X (gfx1250) — matrix operand order repaired and checked by reading, not compiled, not run on hardware
//
#include <hip/hip_runtime.h>
#include <math.h>

typedef __attribute__((ext_vector_type(16))) _Float16 v16h;
typedef __attribute__((ext_vector_type(8)))  _Float16 v8h;
typedef __attribute__((ext_vector_type(4)))  _Float16 v4h;
typedef __attribute__((ext_vector_type(8)))  float    v8f;

#define WAVEFORM_LEN 2048
#define DIM          1024
#define NUM_OSC      128
#define M_TOTAL      8192         // 16 * 512
#define KSUM         256          // 2 * NUM_OSC (cos part | sin part)
#define PI_F         3.14159265358979f

// Workspace layout (in _Float16 units):
//   W16arg : [128][1024]      @ 0
//   W16amp : [128][1024]      @ 131072
//   Bt     : [2048][256]      @ 262144   (Bt[w][k]: k<128 -> cos(pi*z_k*w), else -sin)
//   A2     : [8192][256]      @ 786432   (A2[m][k]: k<128 -> amp*cos(phi),  else amp*sin(phi))

// ---------------------------------------------------------------- prep: W -> f16
__global__ __launch_bounds__(256) void prep_w_kernel(
    const float* __restrict__ W_arg, const float* __restrict__ W_amp,
    _Float16* __restrict__ W16arg, _Float16* __restrict__ W16amp)
{
    int idx = blockIdx.x * 256 + threadIdx.x;          // [0, 131072)
    W16arg[idx] = (_Float16)W_arg[idx];
    W16amp[idx] = (_Float16)W_amp[idx];
}

// ------------------------------------------------- prep: phase table (transposed)
__global__ __launch_bounds__(256) void prep_phase_kernel(
    const float* __restrict__ z_arg, _Float16* __restrict__ Bt)
{
    int idx = blockIdx.x * 256 + threadIdx.x;          // [0, 2048*256)
    int w   = idx >> 8;                                // waveform sample 0..2047
    int kk  = idx & 255;                               // stacked K index 0..255
    int k   = kk & 127;                                // oscillator index
    float ph = PI_F * z_arg[k] * (float)w;
    float s, c;
    __sincosf(ph, &s, &c);
    Bt[idx] = (_Float16)((kk < NUM_OSC) ? c : -s);
}

// ------------------------------------- stage 1: x @ [W_arg|W_amp].T, fused sin/cos
// grid.x = 512 (one 16-row M tile per block); 256 threads = 8 waves,
// wave w owns oscillator columns [16w, 16w+16).
// Double-buffered LDS x-tile (16 x 64 f16 per buffer), one barrier per 64-K step.
__global__ __launch_bounds__(256) void stage1_kernel(
    const float*    __restrict__ x,
    const float*    __restrict__ b_arg,
    const float*    __restrict__ b_amp,
    const _Float16* __restrict__ W16arg,
    const _Float16* __restrict__ W16amp,
    _Float16*       __restrict__ A2)
{
    __shared__ __align__(32) _Float16 xs[2][16 * 64];

    const int tid   = threadIdx.x;
    const int lane  = tid & 31;
    const int wv    = tid >> 5;                        // 0..7
    const int h     = (lane >> 4) & 1;                 // half-wave select
    const int mlane = lane & 15;
    const int m0    = blockIdx.x * 16;
    const int n     = wv * 16 + mlane;                 // oscillator column 0..127

    v8f accArg = {};
    v8f accAmp = {};

    // Loader mapping: 16 rows x 64 cols f32 per tile; 256 threads x 4 f32 each.
    const int ldr = tid >> 4;                          // row 0..15
    const int ldc = (tid & 15) * 4;                    // col 0,4,...,60
    const float* xrow = x + (size_t)(m0 + ldr) * DIM + ldc;

    // Prologue: stage tile 0 into buffer 0.
    {
        const float4 v = *(const float4*)(xrow);
        *(v4h*)(&xs[0][ldr * 64 + ldc]) =
            (v4h){(_Float16)v.x, (_Float16)v.y, (_Float16)v.z, (_Float16)v.w};
    }
    __syncthreads();

    for (int it = 0; it < 16; ++it) {
        const int k0 = it * 64;

        // Issue next tile's global load early (uniform branch: EXEC stays full).
        float4 vnext;
        if (it + 1 < 16) vnext = *(const float4*)(xrow + k0 + 64);

        const _Float16* xb = &xs[it & 1][0];

#pragma unroll
        for (int s = 0; s < 2; ++s) {
            const int ks = s * 32;

            // A fragment: K = 16h + e contiguous along the x row
            v16h a = *(const v16h*)(xb + mlane * 64 + ks + h * 16);

            // B fragments: K = 16h + e contiguous along the W16 row for column n
            v16h ba = *(const v16h*)(W16arg + (size_t)n * DIM + k0 + ks + h * 16);
            v16h bb = *(const v16h*)(W16amp + (size_t)n * DIM + k0 + ks + h * 16);

            accArg = __builtin_amdgcn_wmma_f32_16x16x32_f16(
                false, a, false, ba, (short)0, accArg, false, false);
            accAmp = __builtin_amdgcn_wmma_f32_16x16x32_f16(
                false, a, false, bb, (short)0, accAmp, false, false);
        }

        // One barrier per iteration: separates this iter's reads of buf[it&1]
        // from its rewrite next iteration; no barrier orders the write below before its reads.
        __syncthreads();

        if (it + 1 < 16) {
            *(v4h*)(&xs[(it + 1) & 1][ldr * 64 + ldc]) =
                (v4h){(_Float16)vnext.x, (_Float16)vnext.y,
                      (_Float16)vnext.z, (_Float16)vnext.w};
        }
    }

    // Epilogue: bias, phi = pi*raw, write A2[m][n] = amp*cos, A2[m][n+128] = amp*sin
    const float ba = b_arg[n];
    const float bb = b_amp[n];
#pragma unroll
    for (int r = 0; r < 8; ++r) {
        const int   m   = m0 + r + 8 * h;              // D layout: row = r + 8h, col = n
        const float phi = (accArg[r] + ba) * PI_F;
        const float amp = accAmp[r] + bb;
        float s, c;
        __sincosf(phi, &s, &c);
        A2[(size_t)m * KSUM + n]           = (_Float16)(amp * c);
        A2[(size_t)m * KSUM + NUM_OSC + n] = (_Float16)(amp * s);
    }
}

// ------------- stage 2: out = (A2 @ Bt^T) / 128  (M=8192, N=2048, K=256)
// 4x4 register-blocked: each wave computes a 64x64 output strip (16 WMMA / K-step).
// Strips: 128 (M) x 32 (N) = 4096 waves = 512 blocks * 8 waves.
__global__ __launch_bounds__(256) void stage2_kernel(
    const _Float16* __restrict__ A2,
    const _Float16* __restrict__ Bt,
    float*          __restrict__ out)
{
    const int tid   = threadIdx.x;
    const int lane  = tid & 31;
    const int wv    = tid >> 5;
    const int h     = (lane >> 4) & 1;
    const int nl    = lane & 15;

    const int strip   = blockIdx.x * 8 + wv;           // 0..4095
    const int strip_m = strip >> 5;                    // 0..127
    const int strip_n = strip & 31;                    // 0..31
    const int m0      = strip_m * 64;
    const int n0      = strip_n * 64;

    v8f acc[4][4];
#pragma unroll
    for (int i = 0; i < 4; ++i)
#pragma unroll
        for (int j = 0; j < 4; ++j)
            acc[i][j] = (v8f){};

    const _Float16* arow0 = A2 + (size_t)(m0 + nl) * KSUM;   // A lane row = lane&15
    const _Float16* brow0 = Bt + (size_t)(n0 + nl) * KSUM;   // B lane col = lane&15

#pragma unroll
    for (int k0 = 0; k0 < KSUM; k0 += 32) {
        v16h a[4];
#pragma unroll
        for (int i = 0; i < 4; ++i) {
            const _Float16* ar = arow0 + (size_t)(i * 16) * KSUM;
            a[i] = *(const v16h*)(ar + k0 + h * 16);
        }

        v16h b[4];
#pragma unroll
        for (int j = 0; j < 4; ++j) {
            b[j] = *(const v16h*)(brow0 + (size_t)(j * 16) * KSUM + k0 + h * 16);
        }

#pragma unroll
        for (int i = 0; i < 4; ++i)
#pragma unroll
            for (int j = 0; j < 4; ++j)
                acc[i][j] = __builtin_amdgcn_wmma_f32_16x16x32_f16(
                    false, a[i], false, b[j], (short)0, acc[i][j], false, false);
    }

    const float scale = 1.0f / (float)NUM_OSC;
#pragma unroll
    for (int i = 0; i < 4; ++i) {
#pragma unroll
        for (int r = 0; r < 8; ++r) {
            const size_t row = (size_t)(m0 + i * 16 + r + 8 * h) * WAVEFORM_LEN;
#pragma unroll
            for (int j = 0; j < 4; ++j) {
                out[row + n0 + j * 16 + nl] = acc[i][j][r] * scale;
            }
        }
    }
}

// ---------------------------------------------------------------------- launcher
extern "C" void kernel_launch(void* const* d_in, const int* in_sizes, int n_in,
                              void* d_out, int out_size, void* d_ws, size_t ws_size,
                              hipStream_t stream) {
    (void)in_sizes; (void)n_in; (void)out_size; (void)ws_size;

    const float* x     = (const float*)d_in[0];
    const float* z_arg = (const float*)d_in[1];
    const float* W_arg = (const float*)d_in[2];
    const float* b_arg = (const float*)d_in[3];
    const float* W_amp = (const float*)d_in[4];
    const float* b_amp = (const float*)d_in[5];
    float* out = (float*)d_out;

    _Float16* ws     = (_Float16*)d_ws;
    _Float16* W16arg = ws;
    _Float16* W16amp = ws + 131072;                    // 128*1024
    _Float16* Bt     = ws + 262144;                    // + 128*1024
    _Float16* A2     = ws + 786432;                    // + 2048*256

    prep_w_kernel    <<<512,  256, 0, stream>>>(W_arg, W_amp, W16arg, W16amp);
    prep_phase_kernel<<<2048, 256, 0, stream>>>(z_arg, Bt);
    stage1_kernel    <<<512,  256, 0, stream>>>(x, b_arg, b_amp, W16arg, W16amp, A2);
    stage2_kernel    <<<512,  256, 0, stream>>>(A2, Bt, out);
}
